// ComplexLSTM_10282151707367
// MI455X (gfx1250) — matrix operand order repaired and checked by reading, not compiled, not run on hardware
//
#include <hip/hip_runtime.h>
#include <stdint.h>

#define BB    64
#define TT    512
#define IN_   512
#define HH    512
#define FOURH 2048
#define KTOT  1024   // IN_ + HH, concatenated K
#define EPSF  1e-8f

typedef __attribute__((ext_vector_type(16))) __bf16 v16bf;
typedef __attribute__((ext_vector_type(8)))  float  v8f;

union FragB { v16bf v; uint4 q[2]; };

static __device__ __forceinline__ uint16_t f2bf(float f) {
    union { float f; uint32_t u; } c; c.f = f;
    uint32_t u = c.u;
    return (uint16_t)((u + 0x7FFFu + ((u >> 16) & 1u)) >> 16);
}

static __device__ __forceinline__ v8f wmma_bf16(v16bf a, v16bf b, v8f c) {
    // D = A(16x32) x B(32x16) + C, f32 accum, CDNA5 wave32 WMMA
    return __builtin_amdgcn_wmma_f32_16x16x32_bf16(
        false, a, false, b, (short)0, c, false, false);
}

// Load a 16x32 bf16 A-fragment from a row-major bf16 matrix with row stride
// `stride` elements. Per-lane layout (CDNA5): lanes 0-15 row M=lane, K chunks
// [kb,kb+8) and [kb+16,kb+24) with kb=0; lanes 16-31 same rows, kb=8.
static __device__ __forceinline__ v16bf load_a_bf16(const uint16_t* base) {
    FragB f;
    f.q[0] = *(const uint4*)(base);
    f.q[1] = *(const uint4*)(base + 16);
    return f.v;
}

// ---------------- setup: weight concat+bf16, bias fold, state zero ----------
__global__ __launch_bounds__(256)
void clstm_setup(const float* __restrict__ i2h_wr, const float* __restrict__ i2h_wi,
                 const float* __restrict__ i2h_br, const float* __restrict__ i2h_bi,
                 const float* __restrict__ h2h_wr, const float* __restrict__ h2h_wi,
                 const float* __restrict__ h2h_br, const float* __restrict__ h2h_bi,
                 uint16_t* __restrict__ wcat_r, uint16_t* __restrict__ wcat_i,
                 float* __restrict__ br_sum, float* __restrict__ bi_sum,
                 float* __restrict__ cr_st, float* __restrict__ ci_st,
                 uint16_t* __restrict__ hbf_r, uint16_t* __restrict__ hbf_i)
{
    const size_t stride = (size_t)gridDim.x * blockDim.x;
    const size_t tid0   = (size_t)blockIdx.x * blockDim.x + threadIdx.x;

    for (size_t i = tid0; i < (size_t)FOURH * KTOT; i += stride) {
        size_t row = i / KTOT, k = i % KTOT;
        float vr = (k < IN_) ? i2h_wr[row * IN_ + k] : h2h_wr[row * HH + (k - IN_)];
        float vi = (k < IN_) ? i2h_wi[row * IN_ + k] : h2h_wi[row * HH + (k - IN_)];
        wcat_r[i] = f2bf(vr);
        wcat_i[i] = f2bf(vi);
    }
    for (size_t i = tid0; i < FOURH; i += stride) {
        br_sum[i] = i2h_br[i] + h2h_br[i];
        bi_sum[i] = i2h_bi[i] + h2h_bi[i];
    }
    for (size_t i = tid0; i < (size_t)BB * HH; i += stride) {
        cr_st[i] = 0.f; ci_st[i] = 0.f;
        hbf_r[i] = 0;   hbf_i[i] = 0;
    }
}

// ---------------- optional: pre-split x into bf16 r/i planes ----------------
__global__ __launch_bounds__(256)
void clstm_xcvt(const float* __restrict__ x,        // (B,T,IN,2)
                uint16_t* __restrict__ xbf_r,       // (B,T,IN)
                uint16_t* __restrict__ xbf_i)
{
    const size_t stride = (size_t)gridDim.x * blockDim.x;
    const size_t tid0   = (size_t)blockIdx.x * blockDim.x + threadIdx.x;
    const size_t N      = (size_t)BB * TT * IN_;
    for (size_t i = tid0; i < N; i += stride) {
        float2 v = ((const float2*)x)[i];
        xbf_r[i] = f2bf(v.x);
        xbf_i[i] = f2bf(v.y);
    }
}

// ---------------- one fused recurrent timestep ------------------------------
// Grid: 64 blocks = 32 N-tiles x 2 M-halves; 256 threads = 8 waves = 4 gates x 2 M-tiles.
// Each wave owns ONE complex 16x16 output tile (4 f32 accumulators).
template<bool XBF>
__global__ __launch_bounds__(256)
void clstm_step(const float* __restrict__ x,          // (B,T,IN,2) fp32 (XBF=false)
                const uint16_t* __restrict__ xbf_r,   // (B,T,IN) bf16 (XBF=true)
                const uint16_t* __restrict__ xbf_i,
                const uint16_t* __restrict__ wcat_r,  // (4H, 1024) bf16
                const uint16_t* __restrict__ wcat_i,
                const float* __restrict__ br_sum,     // (4H)
                const float* __restrict__ bi_sum,
                float* __restrict__ cr_st, float* __restrict__ ci_st,       // (B,H)
                uint16_t* __restrict__ hbf_r, uint16_t* __restrict__ hbf_i, // (B,H) bf16
                float* __restrict__ out, int t)
{
    __shared__ float prL[4][32][16];
    __shared__ float piL[4][32][16];

    const int tid  = threadIdx.x;
    const int wave = tid >> 5;            // 8 waves (wave32)
    const int lane = tid & 31;
    const int g    = wave >> 1;           // gate 0..3
    const int mtl  = wave & 1;            // local M-tile within M-half
    const int mh   = blockIdx.x & 1;      // M-half 0/1 (rows mh*32..+31)
    const int n0   = (blockIdx.x >> 1) * 16;
    const int col  = lane & 15;
    const int hi   = lane >> 4;           // lane group 0/1
    const int kbA  = hi * 8;              // A-frag K base within 32-chunk
    const int kbB  = hi * 8;              // B-frag K base within 32-chunk
    const int rowA = (mh * 2 + mtl) * 16 + col;   // batch row for A fragment

    // Prefetch next timestep's x slice toward L2 while this step computes
    if (t + 1 < TT) {
        if (XBF) {
            __builtin_prefetch(xbf_r + ((size_t)rowA * TT + (t + 1)) * IN_ + (wave & 1) * 256, 0, 0);
            __builtin_prefetch(xbf_i + ((size_t)rowA * TT + (t + 1)) * IN_ + (wave & 1) * 256, 0, 0);
        } else {
            __builtin_prefetch(x + (((size_t)rowA * TT + (t + 1)) * IN_) * 2 + wave * 128, 0, 0);
        }
    }

    const int wrow = g * HH + n0 + col;   // W_cat row = B-matrix column
    const uint16_t* wr_base = wcat_r + (size_t)wrow * KTOT;
    const uint16_t* wi_base = wcat_i + (size_t)wrow * KTOT;

    const v8f vzero = {0.f,0.f,0.f,0.f,0.f,0.f,0.f,0.f};
    v8f arr = vzero, aii = vzero, ari = vzero, air = vzero;

    // ---- K loop, x part (k in [0,512))
    for (int kc = 0; kc < IN_; kc += 32) {
        FragB fbr, fbi;
        {
            const uint16_t* p = wr_base + kc + kbB;
            fbr.q[0] = *(const uint4*)(p);
            fbr.q[1] = *(const uint4*)(p + 16);
            const uint16_t* q = wi_base + kc + kbB;
            fbi.q[0] = *(const uint4*)(q);
            fbi.q[1] = *(const uint4*)(q + 16);
        }
        v16bf ar, ai;
        if (XBF) {
            const size_t base = ((size_t)rowA * TT + t) * IN_ + kc + kbA;
            ar = load_a_bf16(xbf_r + base);
            ai = load_a_bf16(xbf_i + base);
        } else {
            const float* xr = x + (((size_t)rowA * TT + t) * IN_) * 2;
            #pragma unroll
            for (int c = 0; c < 2; ++c) {
                const int k0 = kc + kbA + c * 16;
                const float4* p = (const float4*)(xr + 2 * k0);
                #pragma unroll
                for (int j = 0; j < 4; ++j) {
                    float4 f = p[j];
                    ar[c * 8 + 2 * j]     = (__bf16)f.x;
                    ai[c * 8 + 2 * j]     = (__bf16)f.y;
                    ar[c * 8 + 2 * j + 1] = (__bf16)f.z;
                    ai[c * 8 + 2 * j + 1] = (__bf16)f.w;
                }
            }
        }
        arr = wmma_bf16(ar, fbr.v, arr); // xr*wr
        aii = wmma_bf16(ai, fbi.v, aii); // xi*wi
        ari = wmma_bf16(ar, fbi.v, ari); // xr*wi
        air = wmma_bf16(ai, fbr.v, air); // xi*wr
    }

    // ---- K loop, h part (k in [512,1024)): bf16 state from previous step
    for (int kc = IN_; kc < KTOT; kc += 32) {
        FragB fbr, fbi;
        {
            const uint16_t* p = wr_base + kc + kbB;
            fbr.q[0] = *(const uint4*)(p);
            fbr.q[1] = *(const uint4*)(p + 16);
            const uint16_t* q = wi_base + kc + kbB;
            fbi.q[0] = *(const uint4*)(q);
            fbi.q[1] = *(const uint4*)(q + 16);
        }
        const size_t base = (size_t)rowA * HH + (kc - IN_) + kbA;
        v16bf ar = load_a_bf16(hbf_r + base);
        v16bf ai = load_a_bf16(hbf_i + base);
        arr = wmma_bf16(ar, fbr.v, arr);
        aii = wmma_bf16(ai, fbi.v, aii);
        ari = wmma_bf16(ar, fbi.v, ari);
        air = wmma_bf16(ai, fbr.v, air);
    }

    // ---- epilogue: pr = rr - ii + b, pi = ri + ir + b  -> LDS
    {
        const float brs = br_sum[wrow];
        const float bis = bi_sum[wrow];
        #pragma unroll
        for (int e = 0; e < 8; ++e) {
            const int mloc = mtl * 16 + e + hi * 8;  // C/D layout: row = e + 8*lanegroup
            prL[g][mloc][col] = arr[e] - aii[e] + brs;
            piL[g][mloc][col] = ari[e] + air[e] + bis;
        }
    }
    __syncthreads();

    // ---- gates, state update, outputs (uniform trip count: 2 per thread)
    for (int idx = tid; idx < 32 * 16; idx += 256) {
        const int mloc = idx >> 4;
        const int m    = mh * 32 + mloc;
        const int c_   = idx & 15;
        const int n    = n0 + c_;

        float gir = prL[0][mloc][c_], gii = piL[0][mloc][c_];
        float gfr = prL[1][mloc][c_], gfi = piL[1][mloc][c_];
        float gor = prL[2][mloc][c_], goi = piL[2][mloc][c_];
        float ggr = prL[3][mloc][c_], ggi = piL[3][mloc][c_];

        auto msig = [](float& r, float& i) {
            float mag = sqrtf(r * r + i * i);
            float s = (1.f / (1.f + __expf(-mag))) / (mag + EPSF);
            r *= s; i *= s;
        };
        auto mtanh = [](float& r, float& i) {
            float mag = sqrtf(r * r + i * i);
            float s = tanhf(mag) / (mag + EPSF);
            r *= s; i *= s;
        };
        msig(gir, gii); msig(gfr, gfi); msig(gor, goi); mtanh(ggr, ggi);

        const size_t sidx = (size_t)m * HH + n;
        float crn = cr_st[sidx] * gfr + gir * ggr;
        float cin = ci_st[sidx] * gfi + gii * ggi;
        float tr = crn, ti = cin;
        mtanh(tr, ti);
        float hrn = gor * tr - goi * ti;
        float hin = gor * ti + goi * tr;

        cr_st[sidx] = crn; ci_st[sidx] = cin;
        hbf_r[sidx] = f2bf(hrn); hbf_i[sidx] = f2bf(hin);

        const size_t yidx = (((size_t)m * TT + t) * HH + n) * 2;
        out[yidx]     = hrn;
        out[yidx + 1] = hin;

        if (t == TT - 1) {
            const size_t hb = (size_t)BB * TT * HH * 2;
            out[hb + sidx * 2]     = hrn;
            out[hb + sidx * 2 + 1] = hin;
            const size_t cb = hb + (size_t)BB * HH * 2;
            out[cb + sidx * 2]     = crn;
            out[cb + sidx * 2 + 1] = cin;
        }
    }
}

// ---------------- host launcher --------------------------------------------
extern "C" void kernel_launch(void* const* d_in, const int* in_sizes, int n_in,
                              void* d_out, int out_size, void* d_ws, size_t ws_size,
                              hipStream_t stream) {
    const float* x       = (const float*)d_in[0];
    const float* i2h_wr  = (const float*)d_in[1];
    const float* i2h_wi  = (const float*)d_in[2];
    const float* i2h_br  = (const float*)d_in[3];
    const float* i2h_bi  = (const float*)d_in[4];
    const float* h2h_wr  = (const float*)d_in[5];
    const float* h2h_wi  = (const float*)d_in[6];
    const float* h2h_br  = (const float*)d_in[7];
    const float* h2h_bi  = (const float*)d_in[8];

    char* ws = (char*)d_ws;
    uint16_t* wcat_r = (uint16_t*)(ws);                         // 4 MB
    uint16_t* wcat_i = (uint16_t*)(ws + 4194304);               // 4 MB
    float*    br_sum = (float*)(ws + 8388608);                  // 8 KB
    float*    bi_sum = (float*)(ws + 8396800);                  // 8 KB
    float*    cr_st  = (float*)(ws + 8404992);                  // 128 KB
    float*    ci_st  = (float*)(ws + 8536064);                  // 128 KB
    uint16_t* hbf_r  = (uint16_t*)(ws + 8667136);               // 64 KB
    uint16_t* hbf_i  = (uint16_t*)(ws + 8732672);               // 64 KB
    // Optional bf16 x planes (aligned at 16 MB): 32 MB each
    const size_t XBF_R_OFF = 16777216;
    const size_t XBF_I_OFF = XBF_R_OFF + (size_t)BB * TT * IN_ * 2;
    const size_t WS_NEED_XBF = XBF_I_OFF + (size_t)BB * TT * IN_ * 2;
    uint16_t* xbf_r = (uint16_t*)(ws + XBF_R_OFF);
    uint16_t* xbf_i = (uint16_t*)(ws + XBF_I_OFF);
    const bool use_xbf = (ws_size >= WS_NEED_XBF);

    clstm_setup<<<2048, 256, 0, stream>>>(
        i2h_wr, i2h_wi, i2h_br, i2h_bi, h2h_wr, h2h_wi, h2h_br, h2h_bi,
        wcat_r, wcat_i, br_sum, bi_sum, cr_st, ci_st, hbf_r, hbf_i);

    if (use_xbf) {
        clstm_xcvt<<<4096, 256, 0, stream>>>(x, xbf_r, xbf_i);
        for (int t = 0; t < TT; ++t) {
            clstm_step<true><<<64, 256, 0, stream>>>(
                x, xbf_r, xbf_i, wcat_r, wcat_i, br_sum, bi_sum,
                cr_st, ci_st, hbf_r, hbf_i, (float*)d_out, t);
        }
    } else {
        for (int t = 0; t < TT; ++t) {
            clstm_step<false><<<64, 256, 0, stream>>>(
                x, nullptr, nullptr, wcat_r, wcat_i, br_sum, bi_sum,
                cr_st, ci_st, hbf_r, hbf_i, (float*)d_out, t);
        }
    }
}
